// MultiheadCausalSelfAttention_36661840839057
// MI455X (gfx1250) — matrix operand order repaired and checked by reading, not compiled, not run on hardware
//
#include <hip/hip_runtime.h>
#include <hip/hip_bf16.h>

// ---------------------------------------------------------------------------
// Multihead causal self-attention forward on gfx1250 (MI455X).
//   - bf16 WMMA (v_wmma_f32_16x16x32_bf16) for QKV GEMM, QK^T, PV, proj GEMM
//   - Tensor Data Mover (tensor_load_to_lds) double-buffered LDS staging
//   - s_wait_tensorcnt / s_wait_dscnt explicit dependency management
// ---------------------------------------------------------------------------

typedef __bf16 bf16;
typedef __attribute__((ext_vector_type(16))) __bf16 v16bf;
typedef __attribute__((ext_vector_type(8)))  __bf16 v8bf;
typedef __attribute__((ext_vector_type(8)))  float  v8f;
typedef __attribute__((ext_vector_type(4)))  unsigned int v4u;
typedef __attribute__((ext_vector_type(4)))  int v4i;
typedef __attribute__((ext_vector_type(8)))  int v8i;

constexpr int NB  = 2;
constexpr int NT  = 2048;
constexpr int NC  = 1024;
constexpr int NH  = 16;
constexpr int ND  = 64;
constexpr int N3C = 3 * NC;
constexpr int NM  = NB * NT;   // 4096 flattened (b,t) rows

// ---- helpers --------------------------------------------------------------

__device__ __forceinline__ unsigned short f2bf_u(float f) {
  unsigned int u = __float_as_uint(f);
  u += 0x7FFFu + ((u >> 16) & 1u);   // round-to-nearest-even
  return (unsigned short)(u >> 16);
}

__device__ __forceinline__ v8f zero_v8f() {
  v8f z;
#pragma unroll
  for (int i = 0; i < 8; ++i) z[i] = 0.0f;
  return z;
}

__device__ __forceinline__ unsigned rfl(unsigned v) {
  return (unsigned)__builtin_amdgcn_readfirstlane((int)v);
}

// A fragment: 16x32 (MxK) bf16 tile from a row-major matrix, leading dim ld.
// ISA layout: lane row = lane&15 ; k = (j&7) + 8*(lane>>4) + 16*(j>>3)
__device__ __forceinline__ v16bf load_a_frag(const bf16* base, int ld) {
  const int lane = threadIdx.x & 31;
  const int lo = lane & 15, hi = lane >> 4;
  const bf16* p = base + lo * ld + hi * 8;
  v8bf a0 = *(const v8bf*)(p);
  v8bf a1 = *(const v8bf*)(p + 16);
  v16bf a;
#pragma unroll
  for (int j = 0; j < 8; ++j) { a[j] = a0[j]; a[j + 8] = a1[j]; }
  return a;
}

// B fragment: 32x16 (KxN) bf16, stored K-contiguous: element (k,n) at
// baseT[n*ld + k].  ISA layout: lane col = lane&15 ; k = (j&7) + 8*(lane>>4) + 16*(j>>3)
__device__ __forceinline__ v16bf load_bt_frag(const bf16* baseT, int ld) {
  const int lane = threadIdx.x & 31;
  const int lo = lane & 15, hi = lane >> 4;
  const bf16* p = baseT + lo * ld + hi * 8;
  v8bf b0 = *(const v8bf*)(p);
  v8bf b1 = *(const v8bf*)(p + 16);
  v16bf b;
#pragma unroll
  for (int j = 0; j < 8; ++j) { b[j] = b0[j]; b[j + 8] = b1[j]; }
  return b;
}

__device__ __forceinline__ v8f wmma_bf16(v16bf a, v16bf b, v8f c) {
  return __builtin_amdgcn_wmma_f32_16x16x32_bf16(
      /*neg_a=*/false, a, /*neg_b=*/false, b,
      /*c_mod=*/(short)0, c, /*reuse_a=*/false, /*reuse_b=*/false);
}

// ---- Tensor Data Mover: 2-D bf16 tile  global -> LDS ----------------------
// D# per CDNA5 ISA: group0 {count,lds,global,type=2},
// group1 {data_size=2B, tensor dims (OOB bound only), tile dims, row stride}.
// LDS byte offset = low 32 bits of the generic shared-aperture address.
__device__ __forceinline__ void tdm_load_tile_2d(const void* lds_ptr,
                                                 const void* gsrc,
                                                 unsigned tile_d0,      // contiguous elems
                                                 unsigned tile_d1,      // rows
                                                 unsigned stride_elems, // row stride
                                                 unsigned tensor_d0,
                                                 unsigned tensor_d1) {
  const unsigned lds_off = rfl((unsigned)(unsigned long long)lds_ptr);
  const unsigned long long ga = (unsigned long long)gsrc;
  const unsigned ga_lo = rfl((unsigned)ga);
  const unsigned ga_hi = rfl((unsigned)(ga >> 32));

  v4u g0;
  g0[0] = 1u;                                    // one valid descriptor
  g0[1] = lds_off;                               // lds_addr (bytes)
  g0[2] = ga_lo;                                 // global_addr[31:0]
  g0[3] = (ga_hi & 0x01FFFFFFu) | (2u << 30);    // global_addr[56:32] | type=2

  v8i g1;
  g1[0] = (int)(1u << 16);                       // data_size = 2 bytes
  g1[1] = (int)((tensor_d0 & 0xFFFFu) << 16);    // tensor_dim0[15:0]
  g1[2] = (int)(((tensor_d0 >> 16) & 0xFFFFu) | ((tensor_d1 & 0xFFFFu) << 16));
  g1[3] = (int)(((tensor_d1 >> 16) & 0xFFFFu) | ((tile_d0 & 0xFFFFu) << 16));
  g1[4] = (int)(tile_d1 & 0xFFFFu);              // tile_dim1 (tile_dim2 = 0)
  g1[5] = (int)stride_elems;                     // tensor_dim0_stride[31:0]
  g1[6] = 0;
  g1[7] = 0;

  v4i z4 = {0, 0, 0, 0};
#if defined(__clang_major__) && (__clang_major__ >= 23)
  v8i z8 = {0, 0, 0, 0, 0, 0, 0, 0};
  __builtin_amdgcn_tensor_load_to_lds(g0, g1, z4, z4, z8, 0);
#else
  __builtin_amdgcn_tensor_load_to_lds(g0, g1, z4, z4, 0);
#endif
}

// ---- kernel 1: convert fp32 -> bf16, transpose weights --------------------

__global__ void __launch_bounds__(256)
mhsa_cvt_pack(const float* __restrict__ x, const float* __restrict__ w_attn,
              const float* __restrict__ w_proj,
              unsigned short* __restrict__ xb,   // [NM, NC] row-major
              unsigned short* __restrict__ waT,  // [N3C, NC] K-contiguous
              unsigned short* __restrict__ wpT)  // [NC, NC]  K-contiguous
{
  const int id = blockIdx.x * blockDim.x + threadIdx.x;
  const int stride = gridDim.x * blockDim.x;
  for (int i = id; i < NM * NC; i += stride) xb[i] = f2bf_u(x[i]);
  for (int i = id; i < N3C * NC; i += stride) {
    int n = i / NC, k = i - n * NC;
    waT[i] = f2bf_u(w_attn[k * N3C + n]);
  }
  for (int i = id; i < NC * NC; i += stride) {
    int n = i / NC, k = i - n * NC;
    wpT[i] = f2bf_u(w_proj[k * NC + n]);
  }
}

// ---- kernel 2: qkv = x @ w_attn + b_attn ; scatter to Q/K/Vt --------------
// Block = 8 waves = 128(M) x 64(N) tile.  Shared B k-slab (64x32 bf16, 4 KB)
// streamed by TDM into a double-buffered LDS slab; wave 0 drives the DMA.

__global__ void __launch_bounds__(256)
mhsa_qkv_wmma(const unsigned short* __restrict__ xb_,
              const unsigned short* __restrict__ waT_,
              const float* __restrict__ b_attn,
              unsigned short* __restrict__ qbuf,   // [NB*NH, NT, ND]
              unsigned short* __restrict__ kbuf,   // [NB*NH, NT, ND]
              unsigned short* __restrict__ vtbuf)  // [NB*NH, ND, NT]
{
  __shared__ unsigned short ldsB[2][64 * 32];      // [n rows][k] bf16

  const bf16* xb  = (const bf16*)xb_;
  const int wave  = threadIdx.x >> 5;
  const int lane  = threadIdx.x & 31, lo = lane & 15, hi = lane >> 4;
  const int mb    = blockIdx.x & 31;               // 32 M-blocks of 128
  const int nb    = blockIdx.x >> 5;               // 48 N-tiles of 64
  const int m0    = mb * 128 + wave * 16;
  const int n0    = nb * 64;

  v8f acc[4];
#pragma unroll
  for (int t = 0; t < 4; ++t) acc[t] = zero_v8f();

  constexpr int NK = NC / 32;                      // 32 k-steps
  if (wave == 0)
    tdm_load_tile_2d(&ldsB[0][0], waT_ + (size_t)n0 * NC, 32, 64, NC, NC, N3C);

  for (int s = 0; s < NK; ++s) {
    if (wave == 0) __builtin_amdgcn_s_wait_tensorcnt(0);
    __syncthreads();                               // slab s ready; s-1 retired
    if (wave == 0 && s + 1 < NK)
      tdm_load_tile_2d(&ldsB[(s + 1) & 1][0],
                       waT_ + (size_t)n0 * NC + (s + 1) * 32, 32, 64, NC, NC, N3C);

    const bf16* Bs = (const bf16*)ldsB[s & 1];
    v16bf a = load_a_frag(xb + (size_t)m0 * NC + s * 32, NC);
#pragma unroll
    for (int t = 0; t < 4; ++t)
      acc[t] = wmma_bf16(a, load_bt_frag(Bs + 16 * t * 32, 32), acc[t]);
  }

#pragma unroll
  for (int t = 0; t < 4; ++t) {
    const int n   = n0 + 16 * t + lo;              // 0..3071
    const int sel = n / NC;                        // 0=q 1=k 2=v
    const int c   = n - sel * NC;
    const int h   = c >> 6, d = c & 63;
    const float bias = b_attn[n];
#pragma unroll
    for (int r = 0; r < 8; ++r) {
      const int gt = m0 + r + 8 * hi;              // 0..4095
      const int bb = gt >> 11, tt = gt & (NT - 1);
      const int bh = bb * NH + h;
      const unsigned short v = f2bf_u(acc[t][r] + bias);
      if (sel == 0)      qbuf[(bh * NT + tt) * ND + d] = v;
      else if (sel == 1) kbuf[(bh * NT + tt) * ND + d] = v;
      else               vtbuf[(bh * ND + d) * NT + tt] = v;
    }
  }
}

// ---- kernel 3: flash-style causal attention -------------------------------
// blockDim = 64 (2 waves); each wave owns 16 query rows, streams its causal
// K/V tiles through per-wave TDM double buffers (TENSORcnt is per-wave).

__global__ void __launch_bounds__(64)
mhsa_flash_attn(const unsigned short* __restrict__ qbuf_,
                const unsigned short* __restrict__ kbuf_,
                const unsigned short* __restrict__ vtbuf_,
                unsigned short* __restrict__ ybuf)   // [NM, NC] bf16
{
  __shared__ unsigned short ldsK[2][2][32 * 64];   // [wave][buf][key][d]
  __shared__ unsigned short ldsV[2][2][64 * 32];   // [wave][buf][d][key]
  __shared__ unsigned short ldsP[2][16 * 32];      // per-wave P scratch

  const bf16* qbuf = (const bf16*)qbuf_;

  const int wave = threadIdx.x >> 5;
  const int lane = threadIdx.x & 31, lo = lane & 15, hi = lane >> 4;
  const int bh = blockIdx.y;                       // 0..NB*NH-1
  const int b  = bh >> 4, h = bh & (NH - 1);
  const int qtile = blockIdx.x * 2 + wave;         // 0..127
  const int q0 = qtile * 16;

  const bf16* Qh = qbuf + (size_t)bh * NT * ND;
  const unsigned short* Kh = kbuf_ + (size_t)bh * NT * ND;    // [t][d]
  const unsigned short* Vt = vtbuf_ + (size_t)bh * ND * NT;   // [d][t]

  const v16bf aq0 = load_a_frag(Qh + q0 * ND, ND);       // d = 0..31
  const v16bf aq1 = load_a_frag(Qh + q0 * ND + 32, ND);  // d = 32..63

  float m[8], l[8];
  v8f acc[4];
#pragma unroll
  for (int r = 0; r < 8; ++r) { m[r] = -1e30f; l[r] = 0.0f; }
#pragma unroll
  for (int t = 0; t < 4; ++t) acc[t] = zero_v8f();

  const int kbmax = (q0 + 15) >> 5;
  // prologue: stream first K/V tiles
  tdm_load_tile_2d(&ldsK[wave][0][0], Kh, 64, 32, ND, ND, NT);
  tdm_load_tile_2d(&ldsV[wave][0][0], Vt, 32, 64, NT, NT, ND);

  for (int kb = 0; kb <= kbmax; ++kb) {
    const int key0 = kb * 32;
    if (kb < kbmax) {                              // stream next tiles
      tdm_load_tile_2d(&ldsK[wave][(kb + 1) & 1][0],
                       Kh + (size_t)(key0 + 32) * ND, 64, 32, ND, ND, NT);
      tdm_load_tile_2d(&ldsV[wave][(kb + 1) & 1][0],
                       Vt + (size_t)(key0 + 32), 32, 64, NT, NT, ND);
      __builtin_amdgcn_s_wait_tensorcnt(2);        // oldest pair landed
    } else {
      __builtin_amdgcn_s_wait_tensorcnt(0);
    }

    const bf16* Kb = (const bf16*)ldsK[wave][kb & 1];   // [32 key][64 d]
    const bf16* Vb = (const bf16*)ldsV[wave][kb & 1];   // [64 d][32 key]

    // S = Q K^T : two 16x16 C tiles (keys [0,16) and [16,32) of this block)
    v8f s0 = zero_v8f(), s1 = zero_v8f();
    s0 = wmma_bf16(aq0, load_bt_frag(Kb, 64), s0);
    s0 = wmma_bf16(aq1, load_bt_frag(Kb + 32, 64), s0);
    s1 = wmma_bf16(aq0, load_bt_frag(Kb + 16 * 64, 64), s1);
    s1 = wmma_bf16(aq1, load_bt_frag(Kb + 16 * 64 + 32, 64), s1);

    float p0[8], p1[8], ex[8];
#pragma unroll
    for (int r = 0; r < 8; ++r) {
      const int row = q0 + r + 8 * hi;
      float v0 = (key0 + lo      <= row) ? s0[r] * 0.125f : -1e30f;
      float v1 = (key0 + 16 + lo <= row) ? s1[r] * 0.125f : -1e30f;
      float tmax = fmaxf(v0, v1);
      tmax = fmaxf(tmax, __shfl_xor(tmax, 1, 32));
      tmax = fmaxf(tmax, __shfl_xor(tmax, 2, 32));
      tmax = fmaxf(tmax, __shfl_xor(tmax, 4, 32));
      tmax = fmaxf(tmax, __shfl_xor(tmax, 8, 32));
      const float mn = fmaxf(m[r], tmax);
      ex[r] = __expf(m[r] - mn);
      p0[r] = __expf(v0 - mn);
      p1[r] = __expf(v1 - mn);
      float rs = p0[r] + p1[r];
      rs += __shfl_xor(rs, 1, 32);
      rs += __shfl_xor(rs, 2, 32);
      rs += __shfl_xor(rs, 4, 32);
      rs += __shfl_xor(rs, 8, 32);
      l[r] = l[r] * ex[r] + rs;
      m[r] = mn;
    }
#pragma unroll
    for (int t = 0; t < 4; ++t)
#pragma unroll
      for (int r = 0; r < 8; ++r) acc[t][r] *= ex[r];

    // C-layout P -> A-fragment layout via per-wave LDS scratch.
    unsigned short* P = ldsP[wave];
#pragma unroll
    for (int r = 0; r < 8; ++r) {
      const int row = r + 8 * hi;
      P[row * 32 + lo]      = f2bf_u(p0[r]);
      P[row * 32 + lo + 16] = f2bf_u(p1[r]);
    }
    asm volatile("s_wait_dscnt 0" ::: "memory");   // in-wave LDS RAW
    const v16bf pa = load_a_frag((const bf16*)P, 32);

    // O += P V
#pragma unroll
    for (int t = 0; t < 4; ++t)
      acc[t] = wmma_bf16(pa, load_bt_frag(Vb + 16 * t * 32, 32), acc[t]);
  }

  // finalize + store y (bf16 [NM, NC]; col = h*64 + d)
  float rl[8];
#pragma unroll
  for (int r = 0; r < 8; ++r) rl[r] = 1.0f / l[r];
  const int col0 = h * ND;
#pragma unroll
  for (int t = 0; t < 4; ++t)
#pragma unroll
    for (int r = 0; r < 8; ++r) {
      const int gt = b * NT + q0 + r + 8 * hi;
      ybuf[(size_t)gt * NC + col0 + 16 * t + lo] = f2bf_u(acc[t][r] * rl[r]);
    }
}

// ---- kernel 4: out = y @ w_proj + b_proj (fp32 out) -----------------------

__global__ void __launch_bounds__(256)
mhsa_proj_wmma(const unsigned short* __restrict__ yb_,
               const unsigned short* __restrict__ wpT_,
               const float* __restrict__ b_proj,
               float* __restrict__ out)
{
  __shared__ unsigned short ldsB[2][64 * 32];

  const bf16* yb = (const bf16*)yb_;
  const int wave = threadIdx.x >> 5;
  const int lane = threadIdx.x & 31, lo = lane & 15, hi = lane >> 4;
  const int mb   = blockIdx.x & 31;                // 32 M-blocks of 128
  const int nb   = blockIdx.x >> 5;                // 16 N-tiles of 64
  const int m0   = mb * 128 + wave * 16;
  const int n0   = nb * 64;

  v8f acc[4];
#pragma unroll
  for (int t = 0; t < 4; ++t) acc[t] = zero_v8f();

  constexpr int NK = NC / 32;
  if (wave == 0)
    tdm_load_tile_2d(&ldsB[0][0], wpT_ + (size_t)n0 * NC, 32, 64, NC, NC, NC);

  for (int s = 0; s < NK; ++s) {
    if (wave == 0) __builtin_amdgcn_s_wait_tensorcnt(0);
    __syncthreads();
    if (wave == 0 && s + 1 < NK)
      tdm_load_tile_2d(&ldsB[(s + 1) & 1][0],
                       wpT_ + (size_t)n0 * NC + (s + 1) * 32, 32, 64, NC, NC, NC);

    const bf16* Bs = (const bf16*)ldsB[s & 1];
    v16bf a = load_a_frag(yb + (size_t)m0 * NC + s * 32, NC);
#pragma unroll
    for (int t = 0; t < 4; ++t)
      acc[t] = wmma_bf16(a, load_bt_frag(Bs + 16 * t * 32, 32), acc[t]);
  }

#pragma unroll
  for (int t = 0; t < 4; ++t) {
    const int n = n0 + 16 * t + lo;
    const float bias = b_proj[n];
#pragma unroll
    for (int r = 0; r < 8; ++r) {
      const int gt = m0 + r + 8 * hi;
      out[(size_t)gt * NC + n] = acc[t][r] + bias;
    }
  }
}

// ---- launch ---------------------------------------------------------------

extern "C" void kernel_launch(void* const* d_in, const int* in_sizes, int n_in,
                              void* d_out, int out_size, void* d_ws, size_t ws_size,
                              hipStream_t stream) {
  const float* x      = (const float*)d_in[0];
  const float* w_attn = (const float*)d_in[1];
  const float* b_attn = (const float*)d_in[2];
  const float* w_proj = (const float*)d_in[3];
  const float* b_proj = (const float*)d_in[4];
  float* out = (float*)d_out;

  char* ws = (char*)d_ws;
  size_t off = 0;
  unsigned short* xb  = (unsigned short*)(ws + off); off += (size_t)NM  * NC * 2;  // 8 MiB
  unsigned short* waT = (unsigned short*)(ws + off); off += (size_t)N3C * NC * 2;  // 6 MiB
  unsigned short* wpT = (unsigned short*)(ws + off); off += (size_t)NC  * NC * 2;  // 2 MiB
  unsigned short* qb  = (unsigned short*)(ws + off); off += (size_t)NM  * NC * 2;  // 8 MiB
  unsigned short* kb  = (unsigned short*)(ws + off); off += (size_t)NM  * NC * 2;  // 8 MiB
  unsigned short* vt  = (unsigned short*)(ws + off); off += (size_t)NM  * NC * 2;  // 8 MiB
  unsigned short* yb  = (unsigned short*)(ws + off); off += (size_t)NM  * NC * 2;  // 8 MiB
  (void)ws_size; (void)in_sizes; (void)n_in; (void)out_size;

  mhsa_cvt_pack<<<2048, 256, 0, stream>>>(x, w_attn, w_proj, xb, waT, wpT);

  // 32 M-blocks (128 rows) x 48 N-tiles (64 cols)
  mhsa_qkv_wmma<<<1536, 256, 0, stream>>>(xb, waT, b_attn, qb, kb, vt);

  // 64 q-tile groups (2 waves each) x 32 (b,h) pairs
  mhsa_flash_attn<<<dim3(NT / 32, NB * NH), 64, 0, stream>>>(qb, kb, vt, yb);

  // 32 M-blocks x 16 N-tiles
  mhsa_proj_wmma<<<512, 256, 0, stream>>>(yb, wpT, b_proj, out);
}
